// MixedAttention_53730040873705
// MI455X (gfx1250) — matrix operand order repaired and checked by reading, not compiled, not run on hardware
//
#include <hip/hip_runtime.h>

// ---------------------------------------------------------------------------
// MixedAttention for MI455X (gfx1250): bf16 WMMA everywhere, f32 accumulate.
// Round 2: 32-row register blocking in GEMM + flash (2x WMMA per B-fragment),
//          async global->LDS staging (ASYNCcnt) in the trilinear kernel,
//          global prefetch of the next B k-slab in the GEMM loop.
// ---------------------------------------------------------------------------

typedef __bf16 bf16_t;
typedef __attribute__((ext_vector_type(16))) __bf16 v16bf;
typedef __attribute__((ext_vector_type(8)))  __bf16 v8bf;
typedef __attribute__((ext_vector_type(8)))  float  v8f;

#define B_SZ     2
#define S_LEN    2048
#define D_MODEL  1024
#define N_HEADS  8
#define NT_HEADS 4
#define WIN      32
#define IGNORE_V (-1.0e6f)

static __device__ __forceinline__ v8f zero8() {
  v8f z = {0.f, 0.f, 0.f, 0.f, 0.f, 0.f, 0.f, 0.f};
  return z;
}

static __device__ __forceinline__ v8f wmma_bf16(v16bf a, v16bf b, v8f c) {
  // D = A(16x32) x B(32x16) + C(16x16), f32 accumulate
  return __builtin_amdgcn_wmma_f32_16x16x32_bf16(false, a, false, b, (short)0, c, false, false);
}

// LDS byte offset of a generic pointer that points into __shared__ memory
static __device__ __forceinline__ unsigned lds_off_of(const void* p) {
  return (unsigned)(size_t)(__attribute__((address_space(3))) const void*)p;
}

// async 16-byte global -> LDS copy (tracked by ASYNCcnt)
static __device__ __forceinline__ void async_copy_b128(unsigned lds_byte_off, const void* gaddr) {
  asm volatile("global_load_async_to_lds_b128 %0, %1, off"
               :: "v"(lds_byte_off), "v"(gaddr)
               : "memory");
}

// A fragment (16x32 bf16) from row-major [*, lda].
// Contiguous-k layout used here (same in load_b_frag): lanes 0-15 row M=lane, K 0..15; lanes 16-31 row M=lane-16, K 16..31
static __device__ __forceinline__ v16bf load_a_frag(const bf16_t* A, int row0, int k0, int lda, int lane) {
  const int l = lane & 15, hi = (lane >> 4) & 1;
  const bf16_t* p = A + (size_t)(row0 + l) * lda + k0 + hi * 16;
  return *(const v16bf*)p;
}

// A-layout fragment where every lane reads the SAME row (broadcast row, used for la[m])
static __device__ __forceinline__ v16bf load_row_frag(const bf16_t* A, int row, int k0, int lda, int lane) {
  const int hi = (lane >> 4) & 1;
  const bf16_t* p = A + (size_t)row * lda + k0 + hi * 16;
  return *(const v16bf*)p;
}

// B fragment (32x16, KxN) taken from B^T stored row-major [N, ldb].
// Contiguous-k layout used here (same as load_a_frag): lanes 0-15 col N=lane, K k0..k0+15; lanes 16-31 col N=lane-16, K k0+16..k0+31
static __device__ __forceinline__ v16bf load_b_frag(const bf16_t* Bt, int n0, int k0, int ldb, int lane) {
  const int c = lane & 15, kh = (lane >> 4) & 1;
  const bf16_t* p = Bt + (size_t)(n0 + c) * ldb + k0 + kh * 16;
  return *(const v16bf*)p;
}

// reductions within a 16-lane half (C-layout row lives across one half-wave)
static __device__ __forceinline__ float hmax16(float v) {
  v = fmaxf(v, __shfl_xor(v, 1, 32));
  v = fmaxf(v, __shfl_xor(v, 2, 32));
  v = fmaxf(v, __shfl_xor(v, 4, 32));
  v = fmaxf(v, __shfl_xor(v, 8, 32));
  return v;
}
static __device__ __forceinline__ float hsum16(float v) {
  v += __shfl_xor(v, 1, 32);
  v += __shfl_xor(v, 2, 32);
  v += __shfl_xor(v, 4, 32);
  v += __shfl_xor(v, 8, 32);
  return v;
}

// ---------------------------------------------------------------------------
// conversion kernels
// ---------------------------------------------------------------------------
__global__ void f32_to_bf16_kernel(const float* __restrict__ in, bf16_t* __restrict__ out, int n) {
  int i = blockIdx.x * blockDim.x + threadIdx.x;
  if (i < n) out[i] = (bf16_t)in[i];
}

// in: rows x cols (f32) -> out: cols x rows (bf16)
__global__ void transpose_bf16_kernel(const float* __restrict__ in, bf16_t* __restrict__ out,
                                      int rows, int cols) {
  int i = blockIdx.x * blockDim.x + threadIdx.x;
  if (i < rows * cols) {
    int r = i / cols, c = i % cols;
    out[(size_t)c * rows + r] = (bf16_t)in[(size_t)r * cols + c];
  }
}

// ---------------------------------------------------------------------------
// generic WMMA GEMM with fused epilogues; 4 waves x 32 rows = 128-row blocks
// MODE 0: qkv   (bias + rotary on q,k; scatter q/k row-major per head, v transposed)
// MODE 1: abcde (bias + rotary on a,b,c; scatter per tensor/head row-major)
// MODE 2: out   (bias; f32 store)
// A: M x K bf16 row-major, Bt: N x K bf16 row-major (i.e. B transposed)
// ---------------------------------------------------------------------------
template <int MODE>
__global__ __launch_bounds__(128) void gemm_epi_kernel(
    const bf16_t* __restrict__ A, const bf16_t* __restrict__ Bt, int Kdim, int ncols,
    const float* __restrict__ bias, const float* __restrict__ cosT, const float* __restrict__ sinT,
    bf16_t* __restrict__ d0, bf16_t* __restrict__ d1, bf16_t* __restrict__ d2,
    float* __restrict__ df) {
  const int lane = threadIdx.x & 31, wave = threadIdx.x >> 5;
  const int c = lane & 15, hi = (lane >> 4) & 1;
  const int row0 = blockIdx.x * 128 + wave * 32;
  const int col0 = blockIdx.y * 64;

  v8f acc[2][4];
#pragma unroll
  for (int rt = 0; rt < 2; ++rt)
#pragma unroll
    for (int nt = 0; nt < 4; ++nt) acc[rt][nt] = zero8();

  for (int k0 = 0; k0 < Kdim; k0 += 32) {
    const v16bf af0 = load_a_frag(A, row0, k0, Kdim, lane);
    const v16bf af1 = load_a_frag(A, row0 + 16, k0, Kdim, lane);
    if (k0 + 64 < Kdim)  // prefetch next-next B k-slab (speculative, L2)
      __builtin_prefetch(Bt + (size_t)(col0 + c) * Kdim + k0 + 64, 0, 1);
#pragma unroll
    for (int nt = 0; nt < 4; ++nt) {
      const v16bf bf = load_b_frag(Bt, col0 + nt * 16, k0, Kdim, lane);
      acc[0][nt] = wmma_bf16(af0, bf, acc[0][nt]);
      acc[1][nt] = wmma_bf16(af1, bf, acc[1][nt]);
    }
  }

  const size_t TSZ = (size_t)B_SZ * NT_HEADS * S_LEN * 64;  // per-tensor abcde size

#pragma unroll
  for (int rt = 0; rt < 2; ++rt) {
#pragma unroll
    for (int nt = 0; nt < 4; ++nt) {
      const int col = col0 + nt * 16 + c;
#pragma unroll
      for (int r = 0; r < 8; ++r) {
        const int row = row0 + rt * 16 + r + 8 * hi;  // token index in [0, B*S)
        float v = acc[rt][nt][r] + bias[col];
        if (MODE == 2) {
          df[(size_t)row * ncols + col] = v;
          continue;
        }
        const int b = row >> 11, s = row & 2047;
        const float pv = __shfl_xor(v, 1, 32);  // rotary partner (adjacent d = adjacent lane)
        if (MODE == 0) {
          if (col < 1024) {  // rotary on q and k
            const int d = col & 63;
            const float cs = cosT[s * 32 + (d >> 1)], sn = sinT[s * 32 + (d >> 1)];
            v = (d & 1) ? (pv * sn + v * cs) : (v * cs - pv * sn);
          }
          if (col < 512) {                      // q: [b][h][s][d]
            const int h = col >> 6, d = col & 63;
            d0[(((size_t)(b * N_HEADS + h)) * S_LEN + s) * 64 + d] = (bf16_t)v;
          } else if (col < 1024) {              // k: [b][h][s][d]
            const int j = col - 512, h = j >> 6, d = j & 63;
            d1[(((size_t)(b * N_HEADS + h)) * S_LEN + s) * 64 + d] = (bf16_t)v;
          } else {                              // v transposed: [b][h][d][s]
            const int j = col - 1024, h = j >> 6, d = j & 63;
            d2[(((size_t)(b * N_HEADS + h)) * 64 + d) * S_LEN + s] = (bf16_t)v;
          }
        } else {  // MODE 1: abcde, col = tens*256 + h*64 + d
          const int tens = col >> 8, j = col & 255, h = j >> 6, d = j & 63;
          if (tens < 3) {  // rotary on a, b, c
            const float cs = cosT[s * 32 + (d >> 1)], sn = sinT[s * 32 + (d >> 1)];
            v = (d & 1) ? (pv * sn + v * cs) : (v * cs - pv * sn);
          }
          d0[(size_t)tens * TSZ + (((size_t)(b * NT_HEADS + h)) * S_LEN + s) * 64 + d] = (bf16_t)v;
        }
      }
    }
  }
}

// ---------------------------------------------------------------------------
// causal flash attention: one wave per 32 query rows per (b,h)
// Every K/V B-fragment is shared by the two 16-row tiles (2x WMMA per load).
// q,k: [bh][s][64] bf16; vt: [bh][64][s] bf16; zcat cols 0..511
// ---------------------------------------------------------------------------
__global__ __launch_bounds__(32) void flash_attn_kernel(
    const bf16_t* __restrict__ qg, const bf16_t* __restrict__ kg,
    const bf16_t* __restrict__ vtg, bf16_t* __restrict__ zcat) {
  __shared__ __align__(32) bf16_t sP[2 * 16 * 32];
  const int lane = threadIdx.x & 31;
  const int c = lane & 15, hi = (lane >> 4) & 1;
  const int bh = blockIdx.y;                 // b*8+h
  const int q0 = blockIdx.x * 32;
  const bf16_t* Q  = qg  + (size_t)bh * S_LEN * 64;
  const bf16_t* Km = kg  + (size_t)bh * S_LEN * 64;
  const bf16_t* Vt = vtg + (size_t)bh * 64 * S_LEN;

  v16bf qf[2][2];
#pragma unroll
  for (int rt = 0; rt < 2; ++rt) {
    qf[rt][0] = load_a_frag(Q, q0 + rt * 16, 0, 64, lane);
    qf[rt][1] = load_a_frag(Q, q0 + rt * 16, 32, 64, lane);
  }

  float mrow[2][8], lrow[2][8];
  v8f o[2][4];
#pragma unroll
  for (int rt = 0; rt < 2; ++rt) {
#pragma unroll
    for (int r = 0; r < 8; ++r) { mrow[rt][r] = -3.0e38f; lrow[rt][r] = 0.f; }
#pragma unroll
    for (int dt = 0; dt < 4; ++dt) o[rt][dt] = zero8();
  }

  const int jend = q0 + 32;
  for (int j0 = 0; j0 < jend; j0 += 32) {
    v8f st[2][2];
#pragma unroll
    for (int nt = 0; nt < 2; ++nt) {
      const v16bf bk0 = load_b_frag(Km, j0 + nt * 16, 0, 64, lane);
      const v16bf bk1 = load_b_frag(Km, j0 + nt * 16, 32, 64, lane);
      const int key = j0 + nt * 16 + c;
#pragma unroll
      for (int rt = 0; rt < 2; ++rt) {
        v8f acc = zero8();
        acc = wmma_bf16(qf[rt][0], bk0, acc);
        acc = wmma_bf16(qf[rt][1], bk1, acc);
#pragma unroll
        for (int r = 0; r < 8; ++r) {
          float v = acc[r] * 0.125f;            // 1/sqrt(64)
          if (key > q0 + rt * 16 + r + 8 * hi) v = -1.0e9f;
          acc[r] = v;
        }
        st[rt][nt] = acc;
      }
    }
#pragma unroll
    for (int rt = 0; rt < 2; ++rt) {
      float alpha[8];
#pragma unroll
      for (int r = 0; r < 8; ++r) {
        float mv = hmax16(fmaxf(st[rt][0][r], st[rt][1][r]));
        mv = fmaxf(mv, mrow[rt][r]);
        alpha[r] = __expf(mrow[rt][r] - mv);
        mrow[rt][r] = mv;
      }
#pragma unroll
      for (int dt = 0; dt < 4; ++dt)
#pragma unroll
        for (int r = 0; r < 8; ++r) o[rt][dt][r] *= alpha[r];
#pragma unroll
      for (int r = 0; r < 8; ++r) {
        const float p0 = __expf(st[rt][0][r] - mrow[rt][r]);
        const float p1 = __expf(st[rt][1][r] - mrow[rt][r]);
        sP[rt * 512 + (r + 8 * hi) * 32 + c]      = (bf16_t)p0;  // C-layout -> LDS row-major
        sP[rt * 512 + (r + 8 * hi) * 32 + 16 + c] = (bf16_t)p1;
        lrow[rt][r] = lrow[rt][r] * alpha[r] + hsum16(p0 + p1);
      }
    }
    asm volatile("s_wait_dscnt 0x0" ::: "memory");
    const v16bf pf0 = load_a_frag(sP, 0, 0, 32, lane);        // reload as A fragments
    const v16bf pf1 = load_a_frag(sP + 512, 0, 0, 32, lane);
#pragma unroll
    for (int dt = 0; dt < 4; ++dt) {
      const v16bf bv = load_b_frag(Vt, dt * 16, j0, S_LEN, lane);
      o[0][dt] = wmma_bf16(pf0, bv, o[0][dt]);
      o[1][dt] = wmma_bf16(pf1, bv, o[1][dt]);
    }
  }

  const int b = bh >> 3, h = bh & 7;
#pragma unroll
  for (int rt = 0; rt < 2; ++rt)
#pragma unroll
    for (int dt = 0; dt < 4; ++dt)
#pragma unroll
      for (int r = 0; r < 8; ++r) {
        const int srow = q0 + rt * 16 + r + 8 * hi;
        const float v = o[rt][dt][r] / lrow[rt][r];
        zcat[(size_t)(b * S_LEN + srow) * 768 + h * 64 + dt * 16 + c] = (bf16_t)v;
      }
}

// ---------------------------------------------------------------------------
// windowed trilinear attention: one block per (bh, window), 2 waves x 16 rows
// look-around tiles staged with async global->LDS (ASYNCcnt)
// ab5: 5 tensors [tens][b][th][s][64] bf16; zcat cols 512..767
// ---------------------------------------------------------------------------
__global__ __launch_bounds__(64) void tri_attn_kernel(const bf16_t* __restrict__ ab5,
                                                      bf16_t* __restrict__ zcat) {
  __shared__ __align__(32) bf16_t sA[64 * 64];          // la (look-around a), row-major [m][d]
  __shared__ __align__(32) bf16_t sB[64 * 64];          // lb, row-major [l][d]
  __shared__ __align__(32) bf16_t sD[64 * 64];          // ld, row-major [m][d]
  __shared__ __align__(32) bf16_t sET[64 * 64];         // le TRANSPOSED: [d][l]
  __shared__ __align__(32) bf16_t sStage[2][16 * 64];   // per-wave rowL staging

  const int tid = threadIdx.x;
  const int lane = tid & 31, wave = tid >> 5;
  const int c = lane & 15, hi = (lane >> 4) & 1;
  const int w = blockIdx.x;                 // window
  const int bh = blockIdx.y;                // b*4+h
  const size_t TSZ = (size_t)B_SZ * NT_HEADS * S_LEN * 64;
  const size_t headBase = (size_t)bh * S_LEN * 64;
  const int row0 = (w - 1) * WIN;           // global s of look-around row 0 (may be <0)

  {  // cooperative load of look-around slices (zero pad for w==0; pad rows = wave 0 only)
    const int r = tid;                      // 0..63
    const int gs = row0 + r;
    if (gs >= 0) {
      const size_t ro = (size_t)gs * 64;
      const bf16_t* srcA = ab5 + 0 * TSZ + headBase + ro;
      const bf16_t* srcB = ab5 + 1 * TSZ + headBase + ro;
      const bf16_t* srcD = ab5 + 3 * TSZ + headBase + ro;
      const unsigned offA = lds_off_of(sA + r * 64);
      const unsigned offB = lds_off_of(sB + r * 64);
      const unsigned offD = lds_off_of(sD + r * 64);
#pragma unroll
      for (int i = 0; i < 64; i += 8) {
        async_copy_b128(offA + i * 2, srcA + i);
        async_copy_b128(offB + i * 2, srcB + i);
        async_copy_b128(offD + i * 2, srcD + i);
      }
    } else {
      for (int i = 0; i < 64; ++i) {
        sA[r * 64 + i] = (bf16_t)0.f;
        sB[r * 64 + i] = (bf16_t)0.f;
        sD[r * 64 + i] = (bf16_t)0.f;
      }
    }
    // le transposed (scatter -> plain DS stores)
    const bf16_t* srcE = ab5 + 4 * TSZ + headBase + (size_t)(gs < 0 ? 0 : gs) * 64;
    for (int i = 0; i < 64; ++i)
      sET[i * 64 + r] = (gs < 0) ? (bf16_t)0.f : srcE[i];
    asm volatile("s_wait_asynccnt 0x0" ::: "memory");
  }
  __syncthreads();

  // this wave's c fragments (rows wave*16 .. wave*16+15 of the window)
  const bf16_t* Cp = ab5 + 2 * TSZ + headBase + (size_t)(w * WIN) * 64;
  const v16bf cf0 = load_a_frag(Cp, wave * 16, 0, 64, lane);
  const v16bf cf1 = load_a_frag(Cp, wave * 16, 32, 64, lane);

  float mx[8], su[8];
  v8f rowL[4], z1[4];
#pragma unroll
  for (int r = 0; r < 8; ++r) { mx[r] = -3.0e38f; su[r] = 0.f; }
#pragma unroll
  for (int t = 0; t < 4; ++t) { rowL[t] = zero8(); z1[t] = zero8(); }

  for (int m = 0; m < 64; ++m) {
    const int bbm = (m < 32) ? ((w == 0) ? 0 : (row0 + m)) : (w * WIN + m - 32);
    const v16bf laf0 = load_row_frag(sA, m, 0, 64, lane);
    const v16bf laf1 = load_row_frag(sA, m, 32, 64, lane);
    v16bf tf0, tf1;  // A = c (*) la[m], rebuilt in A layout
#pragma unroll
    for (int i = 0; i < 16; ++i) {
      tf0[i] = (bf16_t)((float)cf0[i] * (float)laf0[i]);
      tf1[i] = (bf16_t)((float)cf1[i] * (float)laf1[i]);
    }
    v8f stile[4];
#pragma unroll
    for (int lt = 0; lt < 4; ++lt) {
      v8f acc = zero8();
      acc = wmma_bf16(tf0, load_b_frag(sB, lt * 16, 0, 64, lane), acc);
      acc = wmma_bf16(tf1, load_b_frag(sB, lt * 16, 32, 64, lane), acc);
      const int l = lt * 16 + c;
      const int bbl = (l < 32) ? ((w == 0) ? 0 : (row0 + l)) : (w * WIN + l - 32);
#pragma unroll
      for (int r = 0; r < 8; ++r) {
        const int npos = w * WIN + wave * 16 + r + 8 * hi;
        float v = acc[r];
        if ((npos < bbl) || (bbl <= bbm) || (v == 0.f)) v = IGNORE_V;  // mask + pad-zero kill
        acc[r] = v * (1.0f / 64.0f);
      }
      stile[lt] = acc;
    }
    float al[8], rm[8];
#pragma unroll
    for (int r = 0; r < 8; ++r) {
      float mv = fmaxf(fmaxf(stile[0][r], stile[1][r]), fmaxf(stile[2][r], stile[3][r]));
      mv = hmax16(mv);
      mv = fmaxf(mv, mx[r]);
      al[r] = __expf(mx[r] - mv);
      mx[r] = mv;
    }
#pragma unroll
    for (int r = 0; r < 8; ++r) {
      float psum = 0.f;
#pragma unroll
      for (int lt = 0; lt < 4; ++lt) {
        const float p = __expf(stile[lt][r] - mx[r]);
        rowL[lt][r] = rowL[lt][r] * al[r] + p;
        psum += p;
      }
      psum = hsum16(psum);
      su[r] = su[r] * al[r] + psum;
      rm[r] = psum;  // row-sum over l of current P_m tile
    }
#pragma unroll
    for (int dt = 0; dt < 4; ++dt) {  // z1 += rm (outer) ld[m]
      const float ldv = (float)sD[m * 64 + dt * 16 + c];
#pragma unroll
      for (int r = 0; r < 8; ++r) z1[dt][r] = z1[dt][r] * al[r] + rm[r] * ldv;
    }
  }

  // z2 = rowL @ le   (stage rowL through LDS to convert C-layout -> A-layout)
  bf16_t* stg = sStage[wave];
#pragma unroll
  for (int lt = 0; lt < 4; ++lt)
#pragma unroll
    for (int r = 0; r < 8; ++r)
      stg[(r + 8 * hi) * 64 + lt * 16 + c] = (bf16_t)rowL[lt][r];
  asm volatile("s_wait_dscnt 0x0" ::: "memory");
  const v16bf rf0 = load_a_frag(stg, 0, 0, 64, lane);
  const v16bf rf1 = load_a_frag(stg, 0, 32, 64, lane);

  const int b = bh >> 2, h = bh & 3;
#pragma unroll
  for (int dt = 0; dt < 4; ++dt) {
    v8f z2 = zero8();
    z2 = wmma_bf16(rf0, load_b_frag(sET, dt * 16, 0, 64, lane), z2);
    z2 = wmma_bf16(rf1, load_b_frag(sET, dt * 16, 32, 64, lane), z2);
#pragma unroll
    for (int r = 0; r < 8; ++r) {
      const int srow = w * WIN + wave * 16 + r + 8 * hi;
      const float v = (z1[dt][r] + z2[r]) / su[r];
      zcat[(size_t)(b * S_LEN + srow) * 768 + 512 + h * 64 + dt * 16 + c] = (bf16_t)v;
    }
  }
}

// ---------------------------------------------------------------------------
// launcher
// ---------------------------------------------------------------------------
extern "C" void kernel_launch(void* const* d_in, const int* in_sizes, int n_in,
                              void* d_out, int out_size, void* d_ws, size_t ws_size,
                              hipStream_t stream) {
  (void)in_sizes; (void)n_in; (void)out_size; (void)ws_size;
  const float* x    = (const float*)d_in[0];
  const float* Wqkv = (const float*)d_in[1];
  const float* bqkv = (const float*)d_in[2];
  const float* Wab  = (const float*)d_in[3];
  const float* bab  = (const float*)d_in[4];
  const float* Wout = (const float*)d_in[5];
  const float* bout = (const float*)d_in[6];
  const float* cosT = (const float*)d_in[7];
  const float* sinT = (const float*)d_in[8];
  float* out = (float*)d_out;

  char* ws = (char*)d_ws;
  size_t off = 0;
  auto carve = [&](size_t bytes) -> void* {
    void* p = (void*)(ws + off);
    off += (bytes + 255) & ~(size_t)255;
    return p;
  };
  const size_t NTOK = (size_t)B_SZ * S_LEN;                       // 4096
  bf16_t* xb      = (bf16_t*)carve(NTOK * D_MODEL * 2);           // 8 MB
  bf16_t* wqkv_t  = (bf16_t*)carve((size_t)1536 * 1024 * 2);      // 3 MB
  bf16_t* wab_t   = (bf16_t*)carve((size_t)1280 * 1024 * 2);      // 2.5 MB
  bf16_t* wout_t  = (bf16_t*)carve((size_t)1024 * 768 * 2);       // 1.5 MB
  bf16_t* qb      = (bf16_t*)carve((size_t)B_SZ * N_HEADS * S_LEN * 64 * 2);
  bf16_t* kb      = (bf16_t*)carve((size_t)B_SZ * N_HEADS * S_LEN * 64 * 2);
  bf16_t* vt      = (bf16_t*)carve((size_t)B_SZ * N_HEADS * S_LEN * 64 * 2);
  bf16_t* ab5     = (bf16_t*)carve((size_t)5 * B_SZ * NT_HEADS * S_LEN * 64 * 2);
  bf16_t* zcat    = (bf16_t*)carve(NTOK * 768 * 2);               // 6 MB

  // conversions
  {
    const int n = (int)(NTOK * D_MODEL);
    f32_to_bf16_kernel<<<(n + 255) / 256, 256, 0, stream>>>(x, xb, n);
    int n1 = 1024 * 1536;
    transpose_bf16_kernel<<<(n1 + 255) / 256, 256, 0, stream>>>(Wqkv, wqkv_t, 1024, 1536);
    int n2 = 1024 * 1280;
    transpose_bf16_kernel<<<(n2 + 255) / 256, 256, 0, stream>>>(Wab, wab_t, 1024, 1280);
    int n3 = 768 * 1024;
    transpose_bf16_kernel<<<(n3 + 255) / 256, 256, 0, stream>>>(Wout, wout_t, 768, 1024);
  }

  // qkv projection (+bias, rotary, scatter)
  {
    dim3 g(32, 24);  // M=4096/128, N=1536/64
    gemm_epi_kernel<0><<<g, 128, 0, stream>>>(xb, wqkv_t, 1024, 1536, bqkv, cosT, sinT,
                                              qb, kb, vt, nullptr);
  }
  // abcde projection (+bias, rotary on a,b,c)
  {
    dim3 g(32, 20);  // N=1280/64
    gemm_epi_kernel<1><<<g, 128, 0, stream>>>(xb, wab_t, 1024, 1280, bab, cosT, sinT,
                                              ab5, nullptr, nullptr, nullptr);
  }
  // causal SDPA -> zcat[:, 0:512]
  {
    dim3 g(S_LEN / 32, B_SZ * N_HEADS);
    flash_attn_kernel<<<g, 32, 0, stream>>>(qb, kb, vt, zcat);
  }
  // windowed trilinear -> zcat[:, 512:768]
  {
    dim3 g(S_LEN / WIN, B_SZ * NT_HEADS);
    tri_attn_kernel<<<g, 64, 0, stream>>>(ab5, zcat);
  }
  // out projection (f32)
  {
    dim3 g(32, 16);  // N=1024/64, K=768
    gemm_epi_kernel<2><<<g, 128, 0, stream>>>(zcat, wout_t, 768, 1024, bout, nullptr, nullptr,
                                              nullptr, nullptr, nullptr, out);
  }
}
